// SuperPoint_18476949307428
// MI455X (gfx1250) — matrix operand order repaired and checked by reading, not compiled, not run on hardware
//
#include <hip/hip_runtime.h>
#include <hip/hip_bf16.h>
#include <math.h>
#include <type_traits>

typedef __attribute__((ext_vector_type(16))) _Float16 v16h;
typedef __attribute__((ext_vector_type(8)))  float    v8f;

#define KCHUNK  256          // K-elements of weights staged in LDS per iteration
#define WSTRIDE (KCHUNK + 8) // padded row stride (f16) -> conflict-free ds_load_b128

// ---------------------------------------------------------------------------
// Fused conv (SAME padding) + BN (y = conv*g + b) + optional ReLU, via WMMA.
// Each wave: 16 output channels x 16 pixels. Block = 4 waves = 64 pixels.
// grid: (ceil(HW/64), ceil(Cout/16), N)
//
// PERM=true: K reduction runs kernel-position-major (k' = kk*Cin + ci),
// Cin a power of two (multiple of 32 for KS=3).  Within a 32-wide WMMA
// fragment the kernel position kk is uniform (scalar shift) and the channel
// index is consecutive: bounds test + clamp once per fragment, addr += HW per
// element.  Weight staging applies the same permutation, so the result is
// unchanged.
// PERM=false: specialized Cin==1 path (first layer only, tail chunk only).
// ---------------------------------------------------------------------------
template<int KS, bool PERM>
__global__ __launch_bounds__(128)
void conv_bn_wmma(const float* __restrict__ in, const float* __restrict__ wgt,
                  const float* __restrict__ gam, const float* __restrict__ bet,
                  float* __restrict__ out, int Cin, int Cout, int H, int W,
                  int relu, int ciShift)
{
    constexpr int KK  = KS * KS;
    constexpr int PAD = KS / 2;

    __shared__ _Float16 wlds[16 * WSTRIDE];   // ~8.25 KB

    const int lane  = threadIdx.x & 31;
    const int wave  = threadIdx.x >> 5;
    const int hlf   = lane >> 4;              // wave32 half (lanes 0-15 / 16-31)
    const int l16   = lane & 15;
    const int n     = blockIdx.z;
    const int coB   = blockIdx.y << 4;
    const int pixB  = (blockIdx.x << 6) + (wave << 4);
    const int HW    = H * W;
    const int Ktot  = Cin * KK;
    const int ciMask = (1 << ciShift) - 1;

    const int  pix   = pixB + l16;
    const bool pixOK = pix < HW;
    const int  py    = pixOK ? (pix / W) : 0;
    const int  px    = pixOK ? (pix % W) : 0;

    const float* basep   = in + (size_t)n * Cin * HW;
    const int    maxAddr = Cin * HW - 1;

    v8f acc = {};

    // one 32x16 f16 B fragment at K' index kbase (+16*hlf)
    auto build_b = [&](int kbase, auto FULLC) -> v16h {
        constexpr bool FULL = decltype(FULLC)::value;
        v16h bm;
        const int k = kbase + (hlf << 4);
        float tmp[16];
        if (PERM) {
            // ---- fragment-uniform decomposition --------------------------
            int kk, ci0;
            if (KS == 1) { kk = 0; ci0 = k; }
            else         { kk = kbase >> ciShift; ci0 = k & ciMask; }
            int ry = (KS == 3) ? (kk / 3) : 0;        // scalar, kk <= 8
            int rx = (KS == 3) ? (kk - ry * 3) : 0;
            int oy = py + ry - PAD;
            int ox = px + rx - PAD;
            bool okf = pixOK & (oy >= 0) & (oy < H) & (ox >= 0) & (ox < W);
            int oyc = oy < 0 ? 0 : (oy >= H ? H - 1 : oy);
            int oxc = ox < 0 ? 0 : (ox >= W ? W - 1 : ox);
            int addr = ci0 * HW + oyc * W + oxc;
            #pragma unroll
            for (int e = 0; e < 16; ++e) {
                int a = addr + e * HW;
                if (!FULL) a = a < maxAddr ? a : maxAddr;
                tmp[e] = basep[a];
            }
            #pragma unroll
            for (int e = 0; e < 16; ++e) {
                bool ok = okf & (FULL || ((k + e) < Ktot));
                bm[e] = (_Float16)(ok ? tmp[e] : 0.0f);
            }
        } else {
            // ---- Cin==1 path (first layer): ci = 0, r = ke ---------------
            bool oks[16];
            #pragma unroll
            for (int e = 0; e < 16; ++e) {
                const int ke = k + e;            // == r since KK == Ktot/1
                int ry = ke / KS;                // div by 3 only
                int rx = ke - ry * KS;
                int oy = py + ry - PAD;
                int ox = px + rx - PAD;
                oks[e] = (ke < Ktot) & pixOK &
                         (oy >= 0) & (oy < H) & (ox >= 0) & (ox < W);
                int oyc = oy < 0 ? 0 : (oy >= H ? H - 1 : oy);
                int oxc = ox < 0 ? 0 : (ox >= W ? W - 1 : ox);
                tmp[e] = basep[oyc * W + oxc];   // ci==0, always in-bounds
            }
            #pragma unroll
            for (int e = 0; e < 16; ++e)
                bm[e] = (_Float16)(oks[e] ? tmp[e] : 0.0f);
        }
        return bm;
    };

    // one 16x32 f16 A fragment from LDS at chunk-local offset ksub
    auto build_a = [&](int ksub) -> v16h {
        v16h a;
        #pragma unroll
        for (int e = 0; e < 16; ++e) {
            int ka = ksub + e + (hlf << 4);
            a[e] = wlds[l16 * WSTRIDE + ka];
        }
        return a;
    };

    for (int kc = 0; kc < Ktot; kc += KCHUNK) {
        // ---- stage 16 x KCHUNK weight tile (f32 -> f16, permuted K) ------
        for (int t = threadIdx.x; t < 16 * KCHUNK; t += blockDim.x) {
            int row = t >> 8;            // KCHUNK == 256
            int kk  = t & 255;
            int k   = kc + kk;
            int co  = coB + row;
            int widx;
            if (PERM) {
                int ci   = k & ciMask;
                int kpos = k >> ciShift;
                widx = ci * KK + kpos;   // permuted K' -> original layout
            } else {
                widx = k;
            }
            float v = (co < Cout && k < Ktot) ? wgt[(size_t)co * Ktot + widx] : 0.0f;
            wlds[row * WSTRIDE + kk] = (_Float16)v;
        }
        if (kc + KCHUNK < Ktot)
            __builtin_prefetch(wgt + (size_t)coB * Ktot + kc + KCHUNK, 0, 0);
        __syncthreads();

        if (PERM && kc + KCHUNK <= Ktot) {
            // full chunk: no K guards, no address clamp (dead for PERM=false)
            for (int ksub = 0; ksub < KCHUNK; ksub += 64) {
                v16h a0 = build_a(ksub);
                v16h b0 = build_b(kc + ksub, std::true_type{});
                acc = __builtin_amdgcn_wmma_f32_16x16x32_f16(
                          false, a0, false, b0, (short)0, acc, false, false);
                v16h a1 = build_a(ksub + 32);
                v16h b1 = build_b(kc + ksub + 32, std::true_type{});
                acc = __builtin_amdgcn_wmma_f32_16x16x32_f16(
                          false, a1, false, b1, (short)0, acc, false, false);
            }
        } else {
            // tail chunk: masked overshoot contributes exactly zero
            int kend = Ktot - kc; if (kend > KCHUNK) kend = KCHUNK;
            for (int ksub = 0; ksub < kend; ksub += 64) {
                v16h a0 = build_a(ksub);
                v16h b0 = build_b(kc + ksub, std::false_type{});
                acc = __builtin_amdgcn_wmma_f32_16x16x32_f16(
                          false, a0, false, b0, (short)0, acc, false, false);
                v16h a1 = build_a(ksub + 32);
                v16h b1 = build_b(kc + ksub + 32, std::false_type{});
                acc = __builtin_amdgcn_wmma_f32_16x16x32_f16(
                          false, a1, false, b1, (short)0, acc, false, false);
            }
        }
        __syncthreads();
    }

    // ---- epilogue: BN + ReLU, store f32 ---------------------------------
    #pragma unroll
    for (int r = 0; r < 8; ++r) {
        int co = coB + r + (hlf << 3);
        if (co < Cout && pixOK) {
            float v = acc[r] * gam[co] + bet[co];
            if (relu) v = fmaxf(v, 0.0f);
            out[(((size_t)n * Cout + co) * HW) + pix] = v;
        }
    }
}

// ---------------------------------------------------------------------------
// 2x2 max pool, stride 2.  NC = N*C planes of HxW input.
// ---------------------------------------------------------------------------
__global__ void maxpool2_k(const float* __restrict__ in, float* __restrict__ out,
                           int NC, int H, int W)
{
    int Ho = H >> 1, Wo = W >> 1;
    int i = blockIdx.x * blockDim.x + threadIdx.x;
    int tot = NC * Ho * Wo;
    if (i >= tot) return;
    int x = i % Wo, y = (i / Wo) % Ho, c = i / (Wo * Ho);
    const float* p = in + ((size_t)c * H + 2 * y) * W + 2 * x;
    out[i] = fmaxf(fmaxf(p[0], p[1]), fmaxf(p[W], p[W + 1]));
}

// ---------------------------------------------------------------------------
// 9x9 max pool SAME (-inf padding == clipped window).  N planes of HxW.
// ---------------------------------------------------------------------------
__global__ void nmspool_k(const float* __restrict__ in, float* __restrict__ out,
                          int N, int H, int W)
{
    int i = blockIdx.x * blockDim.x + threadIdx.x;
    int tot = N * H * W;
    if (i >= tot) return;
    int x = i % W, y = (i / W) % H, n = i / (W * H);
    const float* p = in + (size_t)n * H * W;
    float m = -1e30f;
    int y0 = y - 4; if (y0 < 0) y0 = 0;
    int y1 = y + 4; if (y1 > H - 1) y1 = H - 1;
    int x0 = x - 4; if (x0 < 0) x0 = 0;
    int x1 = x + 4; if (x1 > W - 1) x1 = W - 1;
    for (int yy = y0; yy <= y1; ++yy)
        for (int xx = x0; xx <= x1; ++xx)
            m = fmaxf(m, p[yy * W + xx]);
    out[i] = m;
}

// ---------------------------------------------------------------------------
// Channel softmax over 65, drop last, scatter into 8x-upsampled heat map.
// ---------------------------------------------------------------------------
__global__ void softmax_heat_k(const float* __restrict__ det, float* __restrict__ heat,
                               int N, int h, int w)
{
    int i = blockIdx.x * blockDim.x + threadIdx.x;
    int tot = N * h * w;
    if (i >= tot) return;
    int x = i % w, y = (i / w) % h, n = i / (h * w);
    size_t base = (((size_t)n * 65) * h + y) * w + x;
    size_t stride = (size_t)h * w;
    float mx = -1e30f;
    for (int c = 0; c < 65; ++c) mx = fmaxf(mx, det[base + c * stride]);
    float sum = 0.0f;
    for (int c = 0; c < 65; ++c) sum += __expf(det[base + c * stride] - mx);
    float inv = 1.0f / sum;
    int Hh = h * 8, Ww = w * 8;
    for (int c = 0; c < 64; ++c) {
        float p = __expf(det[base + c * stride] - mx) * inv;
        int cy = c >> 3, cx = c & 7;
        heat[(((size_t)n * Hh) + y * 8 + cy) * Ww + x * 8 + cx] = p;
    }
}

// ---------------------------------------------------------------------------
// In-place L2 normalization along channel dim.  One thread per (n,pos).
// ---------------------------------------------------------------------------
__global__ void l2norm_chan_k(float* __restrict__ d, int N, int C, int HW)
{
    int i = blockIdx.x * blockDim.x + threadIdx.x;
    if (i >= N * HW) return;
    int pos = i % HW, n = i / HW;
    float ss = 0.0f;
    for (int c = 0; c < C; ++c) {
        float v = d[(((size_t)n * C + c) * HW) + pos];
        ss += v * v;
    }
    float inv = 1.0f / fmaxf(sqrtf(ss), 1e-12f);
    for (int c = 0; c < C; ++c)
        d[(((size_t)n * C + c) * HW) + pos] *= inv;
}

// ---------------------------------------------------------------------------
// NMS pointwise helpers
// ---------------------------------------------------------------------------
__global__ void eq_mask_k(const float* a, const float* b, float* m, int n)
{
    int i = blockIdx.x * blockDim.x + threadIdx.x;
    if (i < n) m[i] = (a[i] == b[i]) ? 1.0f : 0.0f;
}
__global__ void supp_scores_k(const float* poolmask, const float* heat,
                              float* out, int n)
{
    int i = blockIdx.x * blockDim.x + threadIdx.x;
    if (i < n) out[i] = (poolmask[i] > 0.0f) ? 0.0f : heat[i];
}
__global__ void mask_update_k(float* mask, const float* poolmask,
                              const float* supps, const float* poolsupp, int n)
{
    int i = blockIdx.x * blockDim.x + threadIdx.x;
    if (i < n) {
        bool old = mask[i] != 0.0f;
        bool nm  = (supps[i] == poolsupp[i]) && !(poolmask[i] > 0.0f);
        mask[i] = (old || nm) ? 1.0f : 0.0f;
    }
}
__global__ void final_scores_k(const float* mask, const float* heat,
                               float* scores, int N, int H, int W)
{
    int i = blockIdx.x * blockDim.x + threadIdx.x;
    if (i >= N * H * W) return;
    int x = i % W, y = (i / W) % H;
    float v = (mask[i] != 0.0f) ? heat[i] : 0.0f;
    if (y < 4 || y >= H - 4 || x < 4 || x >= W - 4) v = -1.0f;
    scores[i] = v;
}

// ---------------------------------------------------------------------------
// Top-512 per image: destructive iterative argmax, one 1024-thread block per
// image, LDS tree reduction each round.
// ---------------------------------------------------------------------------
__global__ __launch_bounds__(1024)
void topk_k(float* __restrict__ scores, float* __restrict__ vals,
            int* __restrict__ idxs, int HW)
{
    __shared__ float bv[1024];
    __shared__ int   bi[1024];
    int n = blockIdx.x;
    float* s = scores + (size_t)n * HW;
    for (int t = 0; t < 512; ++t) {
        float best = -1e30f; int bidx = 0;
        for (int i = threadIdx.x; i < HW; i += blockDim.x) {
            float v = s[i];
            if (v > best) { best = v; bidx = i; }
        }
        bv[threadIdx.x] = best; bi[threadIdx.x] = bidx;
        __syncthreads();
        for (int off = 512; off > 0; off >>= 1) {
            if (threadIdx.x < off && bv[threadIdx.x + off] > bv[threadIdx.x]) {
                bv[threadIdx.x] = bv[threadIdx.x + off];
                bi[threadIdx.x] = bi[threadIdx.x + off];
            }
            __syncthreads();
        }
        if (threadIdx.x == 0) {
            vals[n * 512 + t] = bv[0];
            idxs[n * 512 + t] = bi[0];
            s[bi[0]] = -1e30f;
        }
        __syncthreads();
    }
}

// ---------------------------------------------------------------------------
// Keypoints + scores + bilinear descriptor sampling + L2 norm.
// One wave (32 lanes) per keypoint; each lane owns 8 of 256 channels.
// ---------------------------------------------------------------------------
__global__ void desc_k(const float* __restrict__ dsc, const float* __restrict__ vals,
                       const int* __restrict__ idxs, float* __restrict__ kpts,
                       float* __restrict__ sout, float* __restrict__ dout,
                       int hD, int wD, int Wfull)
{
    int gwave = (blockIdx.x * blockDim.x + threadIdx.x) >> 5;
    int lane  = threadIdx.x & 31;
    int n  = gwave >> 9;
    int kp = gwave & 511;
    if (n >= 2) return;

    float val = vals[n * 512 + kp];
    int   idx = idxs[n * 512 + kp];
    float ky = (float)(idx / Wfull);
    float kx = (float)(idx % Wfull);
    bool valid = val > 0.005f;

    float ix = (kx + 0.5f) * 0.125f - 0.5f;
    float iy = (ky + 0.5f) * 0.125f - 0.5f;
    int x0 = (int)floorf(ix), y0 = (int)floorf(iy);
    float wx = ix - (float)x0, wy = iy - (float)y0;

    float accv[8];
    float ss = 0.0f;
    #pragma unroll
    for (int j = 0; j < 8; ++j) {
        int c = lane + 32 * j;
        const float* plane = dsc + (((size_t)n * 256 + c) * hD) * wD;
        float d4[4];
        #pragma unroll
        for (int q = 0; q < 4; ++q) {
            int yy = y0 + (q >> 1);
            int xx = x0 + (q & 1);
            bool inb = xx >= 0 && xx < wD && yy >= 0 && yy < hD;
            int yc = yy < 0 ? 0 : (yy > hD - 1 ? hD - 1 : yy);
            int xc = xx < 0 ? 0 : (xx > wD - 1 ? wD - 1 : xx);
            float v = plane[yc * wD + xc];
            d4[q] = inb ? v : 0.0f;
        }
        float o = d4[0] * (1 - wy) * (1 - wx) + d4[1] * (1 - wy) * wx
                + d4[2] * wy * (1 - wx)       + d4[3] * wy * wx;
        accv[j] = o;
        ss += o * o;
    }
    #pragma unroll
    for (int m = 16; m > 0; m >>= 1) ss += __shfl_xor(ss, m, 32);
    float inv = 1.0f / fmaxf(sqrtf(ss), 1e-12f);
    float vm = valid ? 1.0f : 0.0f;
    #pragma unroll
    for (int j = 0; j < 8; ++j)
        dout[(((size_t)n * 512 + kp) * 256) + lane + 32 * j] = accv[j] * inv * vm;
    if (lane == 0) {
        kpts[((size_t)n * 512 + kp) * 2 + 0] = kx;
        kpts[((size_t)n * 512 + kp) * 2 + 1] = ky;
        sout[n * 512 + kp] = valid ? val : 0.0f;
    }
}

// ---------------------------------------------------------------------------
// Host orchestration
// ---------------------------------------------------------------------------
extern "C" void kernel_launch(void* const* d_in, const int* in_sizes, int n_in,
                              void* d_out, int out_size, void* d_ws, size_t ws_size,
                              hipStream_t stream)
{
    (void)in_sizes; (void)n_in; (void)out_size; (void)ws_size;

    const float* image = (const float*)d_in[0];
    // conv params, in setup_inputs order:
    // 0 inc1, 1 inc2, 2 d11, 3 d12, 4 d21, 5 d22, 6 d31, 7 d32, 8 pa, 9 pb, 10 da, 11 db
    auto getW = [&](int i) { return (const float*)d_in[1 + 3 * i + 0]; };
    auto getG = [&](int i) { return (const float*)d_in[1 + 3 * i + 1]; };
    auto getB = [&](int i) { return (const float*)d_in[1 + 3 * i + 2]; };

    float* ws   = (float*)d_ws;
    float* BIG0 = ws;
    float* BIG1 = ws + 9830400;            // 2*64*240*320
    float* sp   = ws + 2 * 9830400;
    float* pa_out = sp; sp += 614400;      // 2*256*30*40
    float* da_out = sp; sp += 614400;
    float* det    = sp; sp += 156000;      // 2*65*30*40
    float* dscb   = sp; sp += 614400;      // 2*256*30*40
    float* heat   = sp; sp += 153600;      // 2*240*320
    float* nmsA   = sp; sp += 153600;
    float* nmsB   = sp; sp += 153600;
    float* nmsC   = sp; sp += 153600;
    float* mask   = sp; sp += 153600;
    float* scores = sp; sp += 153600;
    float* vals   = sp; sp += 1024;
    int*   idxs   = (int*)sp;

    auto log2i = [](int v) { int s = 0; while ((1 << s) < v) ++s; return s; };

    // first 3x3 layer (Cin=1): specialized path
    auto conv3_first = [&](const float* in, int ci, int co, int H, int W,
                           int relu, int pi, float* out) {
        dim3 g((H * W + 63) / 64, (co + 15) / 16, 2);
        conv_bn_wmma<3, false><<<g, 128, 0, stream>>>(
            in, getW(pi), getG(pi), getB(pi), out, ci, co, H, W, relu, 0);
    };
    // 3x3 layers with power-of-two Cin >= 32: permuted fast path
    auto conv3 = [&](const float* in, int ci, int co, int H, int W,
                     int relu, int pi, float* out) {
        dim3 g((H * W + 63) / 64, (co + 15) / 16, 2);
        conv_bn_wmma<3, true><<<g, 128, 0, stream>>>(
            in, getW(pi), getG(pi), getB(pi), out, ci, co, H, W, relu, log2i(ci));
    };
    auto conv1 = [&](const float* in, int ci, int co, int H, int W,
                     int relu, int pi, float* out) {
        dim3 g((H * W + 63) / 64, (co + 15) / 16, 2);
        conv_bn_wmma<1, true><<<g, 128, 0, stream>>>(
            in, getW(pi), getG(pi), getB(pi), out, ci, co, H, W, relu, log2i(ci));
    };
    auto pw = [&](int n) { return dim3((n + 255) / 256); };

    // ---- backbone -------------------------------------------------------
    conv3_first(image, 1, 64, 240, 320, 1, 0, BIG0);
    conv3(BIG0, 64, 64, 240, 320, 1, 1, BIG1);
    maxpool2_k<<<pw(2 * 64 * 120 * 160), 256, 0, stream>>>(BIG1, BIG0, 2 * 64, 240, 320);
    conv3(BIG0, 64, 64, 120, 160, 1, 2, BIG1);
    conv3(BIG1, 64, 64, 120, 160, 1, 3, BIG0);
    maxpool2_k<<<pw(2 * 64 * 60 * 80), 256, 0, stream>>>(BIG0, BIG1, 2 * 64, 120, 160);
    conv3(BIG1, 64, 128, 60, 80, 1, 4, BIG0);
    conv3(BIG0, 128, 128, 60, 80, 1, 5, BIG1);
    maxpool2_k<<<pw(2 * 128 * 30 * 40), 256, 0, stream>>>(BIG1, BIG0, 2 * 128, 60, 80);
    conv3(BIG0, 128, 128, 30, 40, 1, 6, BIG1);
    conv3(BIG1, 128, 128, 30, 40, 1, 7, BIG0);   // x = BIG0 (2,128,30,40)

    // ---- heads ----------------------------------------------------------
    conv3(BIG0,   128, 256, 30, 40, 1,  8, pa_out);
    conv1(pa_out, 256,  65, 30, 40, 0,  9, det);
    conv3(BIG0,   128, 256, 30, 40, 1, 10, da_out);
    conv1(da_out, 256, 256, 30, 40, 0, 11, dscb);
    l2norm_chan_k<<<pw(2 * 1200), 256, 0, stream>>>(dscb, 2, 256, 1200);

    // ---- softmax -> heat ------------------------------------------------
    softmax_heat_k<<<pw(2 * 1200), 256, 0, stream>>>(det, heat, 2, 30, 40);

    // ---- NMS ------------------------------------------------------------
    const int HT = 2 * 240 * 320;
    nmspool_k<<<pw(HT), 256, 0, stream>>>(heat, nmsA, 2, 240, 320);
    eq_mask_k<<<pw(HT), 256, 0, stream>>>(heat, nmsA, mask, HT);
    for (int it = 0; it < 2; ++it) {
        nmspool_k<<<pw(HT), 256, 0, stream>>>(mask, nmsA, 2, 240, 320);
        supp_scores_k<<<pw(HT), 256, 0, stream>>>(nmsA, heat, nmsB, HT);
        nmspool_k<<<pw(HT), 256, 0, stream>>>(nmsB, nmsC, 2, 240, 320);
        mask_update_k<<<pw(HT), 256, 0, stream>>>(mask, nmsA, nmsB, nmsC, HT);
    }
    final_scores_k<<<pw(HT), 256, 0, stream>>>(mask, heat, scores, 2, 240, 320);

    // ---- top-k ----------------------------------------------------------
    topk_k<<<2, 1024, 0, stream>>>(scores, vals, idxs, 240 * 320);

    // ---- outputs: kpts (2048) | scores (1024) | descriptors (262144) ----
    float* out   = (float*)d_out;
    float* kpts  = out;
    float* soutp = out + 2048;
    float* doutp = out + 3072;
    desc_k<<<128, 256, 0, stream>>>(dscb, vals, idxs, kpts, soutp, doutp, 30, 40, 320);
}
